// Minerva_37211596652565
// MI455X (gfx1250) — compile-verified, matrix operand order checked by reading, not run on hardware
//
#include <hip/hip_runtime.h>
#include <hip/hip_bf16.h>

typedef unsigned short ushort_t;
typedef __attribute__((ext_vector_type(16))) __bf16       v16bf;
typedef __attribute__((ext_vector_type(8)))  float        v8f;
typedef __attribute__((ext_vector_type(4)))  float        f32x4;
typedef __attribute__((ext_vector_type(2)))  unsigned int u32x2;
typedef __attribute__((ext_vector_type(4)))  unsigned int u32x4;
typedef __attribute__((ext_vector_type(8)))  unsigned int u32x8;
typedef __attribute__((__vector_size__(16))) int          i32x4v;  // async builtin's pointee type

#define DIMD    128   // feature dim d (fixed by reference)
#define KC      64    // M-chunk staged in LDS per iteration
#define BM      128   // X rows per workgroup (8 waves x 16 rows)
#define MSPLIT  4     // M split across blockIdx.y for occupancy

#if __has_builtin(__builtin_amdgcn_global_load_async_to_lds_b128) && \
    __has_builtin(__builtin_amdgcn_s_wait_asynccnt)
#define HAVE_ASYNC_LDS 1
#else
#define HAVE_ASYNC_LDS 0
#endif

// float -> bf16, round-to-nearest-even (prep kernel only; hot loop uses native cvt)
__device__ __forceinline__ ushort_t f2bf(float f) {
  unsigned u = __builtin_bit_cast(unsigned, f);
  unsigned r = u + 0x7FFFu + ((u >> 16) & 1u);
  return (ushort_t)(r >> 16);
}

// 16-element bf16 WMMA fragment: 8 contiguous bf16 at p, 8 contiguous at p+16
// (ISA 16-bit A/B layout: K groups {kbase..kbase+7, kbase+16..kbase+23})
__device__ __forceinline__ v16bf load_frag(const ushort_t* p) {
  u32x8 t;
  ((u32x4*)&t)[0] = *(const u32x4*)(p);
  ((u32x4*)&t)[1] = *(const u32x4*)(p + 16);
  return __builtin_bit_cast(v16bf, t);
}

// Build GEMM2 A-fragment from two cubed a^T accumulators using native f32->bf16
// converts (backend pairs them into v_cvt_pk_bf16_f32).
__device__ __forceinline__ v16bf pack_a3(const v8f& lo, const v8f& hi) {
  v16bf sa;
  #pragma unroll
  for (int j = 0; j < 8; ++j) {
    sa[j]     = (__bf16)lo[j];
    sa[8 + j] = (__bf16)hi[j];
  }
  return sa;
}

#if HAVE_ASYNC_LDS
// One 16-byte async global->LDS copy (tracked by ASYNCcnt).
// Builtin signature (from hipcc diagnostic): (int4 AS1*, int4 AS3*, imm, imm)
__device__ __forceinline__ void async_cp16(const ushort_t* g, ushort_t* l) {
  __builtin_amdgcn_global_load_async_to_lds_b128(
      (__attribute__((address_space(1))) i32x4v*)g,
      (__attribute__((address_space(3))) i32x4v*)l, 0, 0);
}
#endif

// ---------------------------------------------------------------------------
// Prep: one wave32 per row. Normalize X,D rows -> bf16; transpose-convert r so
// the main kernel's B-fragment staging is contiguous (rbT is [d][M] bf16).
// ---------------------------------------------------------------------------
__global__ __launch_bounds__(256) void prep_kernel(
    const float* __restrict__ X, const float* __restrict__ Dm,
    const float* __restrict__ r,
    ushort_t* __restrict__ Xn, ushort_t* __restrict__ Dn,
    ushort_t* __restrict__ rbT, int N, int M)
{
  int gwave = (blockIdx.x * 256 + threadIdx.x) >> 5;
  int lane  = threadIdx.x & 31;
  if (gwave >= N + 2 * M) return;

  if (gwave < N + M) {
    const float* src;
    ushort_t*    dst;
    if (gwave < N) {
      src = X + (size_t)gwave * DIMD;  dst = Xn + (size_t)gwave * DIMD;
    } else {
      int rr = gwave - N;
      src = Dm + (size_t)rr * DIMD;    dst = Dn + (size_t)rr * DIMD;
    }
    f32x4 v = *(const f32x4*)(src + lane * 4);
    float ss = v[0]*v[0] + v[1]*v[1] + v[2]*v[2] + v[3]*v[3];
    #pragma unroll
    for (int off = 16; off > 0; off >>= 1) ss += __shfl_xor(ss, off, 32);
    float scale = 1.0f / fmaxf(sqrtf(ss), 1e-12f);  // x / max(||x||, eps)
    u32x2 o;
    o[0] = (unsigned)f2bf(v[0]*scale) | ((unsigned)f2bf(v[1]*scale) << 16);
    o[1] = (unsigned)f2bf(v[2]*scale) | ((unsigned)f2bf(v[3]*scale) << 16);
    *(u32x2*)(dst + lane * 4) = o;
  } else {
    int rr = gwave - N - M;
    f32x4 v = *(const f32x4*)(r + (size_t)rr * DIMD + lane * 4);
    #pragma unroll
    for (int j = 0; j < 4; ++j)
      rbT[(size_t)(lane * 4 + j) * M + rr] = f2bf(v[j]);
  }
}

// ---------------------------------------------------------------------------
// Stage one KC-chunk: Dn rows (row-major, contiguous 16KB) and r^T rows
// ([d][KC], 128B contiguous per row) -> all-b128 staging, no scatter.
// ---------------------------------------------------------------------------
__device__ __forceinline__ void stage_chunk(
    const ushort_t* __restrict__ Dn, const ushort_t* __restrict__ rbT,
    int m0, int M, ushort_t* dn_buf, ushort_t* rt_buf, int tid)
{
  #pragma unroll
  for (int i = 0; i < 4; ++i) {
    int s   = tid + i * 256;          // 1024 b128 slots per buffer
    int d   = s >> 3;                 // r^T row (d column)
    int seg = (s & 7) * 8;            // 8 bf16 per b128
#if HAVE_ASYNC_LDS
    async_cp16(Dn  + (size_t)m0 * DIMD + s * 8, dn_buf + s * 8);
    async_cp16(rbT + (size_t)d * M + m0 + seg,  rt_buf + d * KC + seg);
#else
    *(u32x4*)(dn_buf + s * 8)       = *(const u32x4*)(Dn  + (size_t)m0 * DIMD + s * 8);
    *(u32x4*)(rt_buf + d * KC + seg)= *(const u32x4*)(rbT + (size_t)d * M + m0 + seg);
#endif
  }
}

// ---------------------------------------------------------------------------
// Fused a^T-GEMM -> cube (registers only) -> echo-GEMM, double-buffered LDS.
// GEMM1 computes a^T = Dn_chunk @ Xn_tile^T so its C/D accumulator layout IS
// the 16-bit A-fragment layout of GEMM2: the cube + relayout never touches LDS.
// ---------------------------------------------------------------------------
__global__ __launch_bounds__(256) void minerva_main(
    const ushort_t* __restrict__ Xn, const ushort_t* __restrict__ Dn,
    const ushort_t* __restrict__ rbT, float* __restrict__ part,
    int N, int M)
{
  __shared__ __align__(16) ushort_t lds_dn[2][KC * DIMD];   // 2 x 16 KB
  __shared__ __align__(16) ushort_t lds_rt[2][DIMD * KC];   // 2 x 16 KB

  const int tid   = threadIdx.x;
  const int wave  = tid >> 5;
  const int lane  = tid & 31;
  const int l16   = lane & 15;
  const int khalf = (lane >> 4) << 3;   // fragment K-half select
  const int mrow  = (lane >> 4) << 3;   // C/D layout row half

  const int row0 = blockIdx.x * BM + wave * 16;

  // Xn tile = B operand of GEMM1 (columns of a^T): resident in VGPRs all kernel
  v16bf xfrag[4];
  #pragma unroll
  for (int kk = 0; kk < 4; ++kk)
    xfrag[kk] = load_frag(Xn + (size_t)(row0 + l16) * DIMD + kk * 32 + khalf);

  v8f zero = {};
  v8f eacc[8];
  #pragma unroll
  for (int i = 0; i < 8; ++i) eacc[i] = zero;

  const int mchunk  = M / MSPLIT;
  const int mlo     = blockIdx.y * mchunk;
  const int nchunks = mchunk / KC;

  stage_chunk(Dn, rbT, mlo, M, lds_dn[0], lds_rt[0], tid);
#if HAVE_ASYNC_LDS
  __builtin_amdgcn_s_wait_asynccnt(0);
#endif
  __syncthreads();

  for (int c = 0; c < nchunks; ++c) {
    const int cur = c & 1;
    const ushort_t* dn_buf = lds_dn[cur];
    const ushort_t* rt_buf = lds_rt[cur];

    // Kick off next chunk's staging into the other buffer (overlaps compute)
    if (c + 1 < nchunks)
      stage_chunk(Dn, rbT, mlo + (c + 1) * KC, M, lds_dn[cur ^ 1], lds_rt[cur ^ 1], tid);
#if !HAVE_ASYNC_LDS
    if (c + 2 < nchunks) {   // L2 prefetch two chunks ahead (global_prefetch_b8)
      const char* nd = (const char*)(Dn + (size_t)(mlo + (c + 2) * KC) * DIMD);
      __builtin_prefetch(nd + tid * 64, 0, 3);
      const char* nr = (const char*)(rbT + (size_t)(tid >> 1) * M + mlo + (c + 2) * KC);
      __builtin_prefetch(nr + (tid & 1) * 64, 0, 3);
    }
#endif

    // --- GEMM1 (transposed): at_kt = Dn[kt-tile] @ Xn_tile^T, then cube ---
    // Issue all 4 fragment loads before the 4 WMMAs so the scheduler can use
    // partial s_wait_dscnt instead of a full drain per WMMA.
    v8f at[4];
    #pragma unroll
    for (int kt = 0; kt < KC / 16; ++kt) {
      v16bf dfrag[4];
      #pragma unroll
      for (int kk = 0; kk < 4; ++kk)
        dfrag[kk] = load_frag(dn_buf + (kt * 16 + l16) * DIMD + kk * 32 + khalf);
      v8f acc = zero;
      #pragma unroll
      for (int kk = 0; kk < 4; ++kk)
        acc = __builtin_amdgcn_wmma_f32_16x16x32_bf16(
                  false, dfrag[kk], false, xfrag[kk], (short)0, acc, false, false);
      #pragma unroll
      for (int e = 0; e < 8; ++e) { float v = acc[e]; acc[e] = v * v * v; }  // sign|a|^3 == a^3
      at[kt] = acc;
    }

    // --- Register-only relayout: a^T accumulators ARE GEMM2 A-fragments ---
    #pragma unroll
    for (int kk = 0; kk < KC / 32; ++kk) {
      v16bf sa = pack_a3(at[2 * kk], at[2 * kk + 1]);   // native v_cvt_pk_bf16_f32
      // --- GEMM2: echo_tile += a^3(16 x 32) @ r_chunk(32 x 128) ---
      // Batch B-fragment loads 4 at a time ahead of their WMMAs (same reason).
      #pragma unroll
      for (int half = 0; half < 2; ++half) {
        v16bf rfrag[4];
        #pragma unroll
        for (int i = 0; i < 4; ++i)
          rfrag[i] = load_frag(rt_buf + ((half * 4 + i) * 16 + l16) * KC + kk * 32 + khalf);
        #pragma unroll
        for (int i = 0; i < 4; ++i)
          eacc[half * 4 + i] = __builtin_amdgcn_wmma_f32_16x16x32_bf16(
              false, sa, false, rfrag[i], (short)0, eacc[half * 4 + i], false, false);
      }
    }

#if HAVE_ASYNC_LDS
    if (c + 1 < nchunks) __builtin_amdgcn_s_wait_asynccnt(0);
#endif
    __syncthreads();
  }

  // Write this split's partial echo (deterministic: one writer per element)
  float* po = part + (size_t)blockIdx.y * N * DIMD;
  #pragma unroll
  for (int nt = 0; nt < 8; ++nt) {
    #pragma unroll
    for (int e = 0; e < 8; ++e) {
      po[(size_t)(row0 + mrow + e) * DIMD + nt * 16 + l16] = eacc[nt][e];
    }
  }
}

// ---------------------------------------------------------------------------
// Deterministic reduction of the MSPLIT partial buffers into d_out.
// ---------------------------------------------------------------------------
__global__ __launch_bounds__(256) void reduce_kernel(
    const float* __restrict__ part, float* __restrict__ out, int nd)
{
  int i = blockIdx.x * 256 + threadIdx.x;
  if (i < nd) {
    float s = 0.0f;
    #pragma unroll
    for (int p = 0; p < MSPLIT; ++p) s += part[(size_t)p * nd + i];
    out[i] = s;
  }
}

extern "C" void kernel_launch(void* const* d_in, const int* in_sizes, int n_in,
                              void* d_out, int out_size, void* d_ws, size_t ws_size,
                              hipStream_t stream) {
  const float* X  = (const float*)d_in[0];
  const float* Dm = (const float*)d_in[1];
  const float* r  = (const float*)d_in[2];
  float* out = (float*)d_out;

  const int N = in_sizes[0] / DIMD;   // 8192
  const int M = in_sizes[1] / DIMD;   // 16384

  // Workspace: Xn(bf16) | Dn(bf16) | rbT(bf16, transposed) | partials(f32 x MSPLIT)
  char* ws = (char*)d_ws;
  ushort_t* Xn  = (ushort_t*)ws;  ws += (size_t)N * DIMD * sizeof(ushort_t);
  ushort_t* Dn  = (ushort_t*)ws;  ws += (size_t)M * DIMD * sizeof(ushort_t);
  ushort_t* rbT = (ushort_t*)ws;  ws += (size_t)M * DIMD * sizeof(ushort_t);
  float* part   = (float*)ws;     // MSPLIT * N * DIMD floats

  int rows    = N + 2 * M;
  int pblocks = (rows + 7) / 8;
  prep_kernel<<<pblocks, 256, 0, stream>>>(X, Dm, r, Xn, Dn, rbT, N, M);

  dim3 grid(N / BM, MSPLIT);
  minerva_main<<<grid, 256, 0, stream>>>(Xn, Dn, rbT, part, N, M);

  int nd = N * DIMD;
  reduce_kernel<<<(nd + 255) / 256, 256, 0, stream>>>(part, out, nd);
}
